// VectorQuantizerEMA_27616639713485
// MI455X (gfx1250) — matrix operand order repaired and checked by reading, not compiled, not run on hardware
//
#include <hip/hip_runtime.h>
#include <math.h>

typedef _Float16 v16h __attribute__((ext_vector_type(16)));
typedef float    v8f  __attribute__((ext_vector_type(8)));
typedef int      v4i  __attribute__((vector_size(16)));

#define D_DIM   256
#define K_CODES 2048
#define N_VEC   32768
#define HW      1024
#define DHW     (D_DIM * HW)        // 262144
#define Q_ELEMS (N_VEC * D_DIM)     // 8388608

#if defined(__HIP_DEVICE_COMPILE__) && \
    __has_builtin(__builtin_amdgcn_global_load_async_to_lds_b128) && \
    __has_builtin(__builtin_amdgcn_s_wait_asynccnt)
#define VQ_ASYNC 1
#else
#define VQ_ASYNC 0
#endif

// ---------------------------------------------------------------------------
// Kernel 1: prep. Per codeword k: zero histogram bin, compute ||e_k||^2,
// repack embedding (D,K) f32 -> (K,D) f16 so each codeword is contiguous
// (enables b128 async staging into LDS and contiguous B-fragment reads).
// ---------------------------------------------------------------------------
__global__ __launch_bounds__(256)
void vq_prep(const float* __restrict__ emb, float* __restrict__ eNorm,
             _Float16* __restrict__ embF16, unsigned* __restrict__ counts) {
    int k = blockIdx.x * blockDim.x + threadIdx.x;
    if (k >= K_CODES) return;
    counts[k] = 0u;
    float s = 0.f;
    #pragma unroll 8
    for (int d = 0; d < D_DIM; ++d) {
        float e = emb[d * K_CODES + k];       // coalesced across k
        s += e * e;
        embF16[(size_t)k * D_DIM + d] = (_Float16)e;
    }
    eNorm[k] = s;
}

#if VQ_ASYNC
typedef __attribute__((address_space(1))) v4i gas1_v4i;
typedef __attribute__((address_space(3))) v4i las3_v4i;

// Stage one 16-codeword tile (8 KB) into LDS with async-to-LDS b128 copies:
// 128 threads x 4 instr x 16 B, no VGPR data movement, tracked by ASYNCcnt.
__device__ __forceinline__ void stage_async(const _Float16* g, _Float16* l, int tid) {
    gas1_v4i* gp = (gas1_v4i*)(uintptr_t)g;
    las3_v4i* lp = (las3_v4i*)(unsigned)(uintptr_t)l;   // low 32 bits = LDS offset
    #pragma unroll
    for (int i = 0; i < 4; ++i)
        __builtin_amdgcn_global_load_async_to_lds_b128(
            gp + tid + 128 * i, lp + tid + 128 * i, 0, 0);
}
#endif

// ---------------------------------------------------------------------------
// Kernel 2: fused distance GEMM + argmin via v_wmma_f32_16x16x32_f16.
// Block = 128 threads (4 waves); each wave owns 32 rows (two 16x256 A tiles
// resident in VGPRs), so every staged B tile feeds 16 WMMAs. 16-codeword
// B tiles are double-buffered in LDS and filled asynchronously while the
// previous tile computes.  dist = ||e||^2 - 2 x.e (row norm dropped).
// ---------------------------------------------------------------------------
__global__ __launch_bounds__(128)
void vq_argmin(const float* __restrict__ inp, const _Float16* __restrict__ embF16,
               const float* __restrict__ eNorm, int* __restrict__ idxOut,
               float* __restrict__ idxOutF, unsigned* __restrict__ counts) {
    __shared__ _Float16 Bsh0[16 * D_DIM];             // 8 KB, buffer 0
    __shared__ _Float16 Bsh1[16 * D_DIM];             // 8 KB, buffer 1

    const int tid  = threadIdx.x;
    const int wave = tid >> 5;
    const int lane = tid & 31;
    const int m    = lane & 15;                       // row (A) / column (B/C)
    const int g    = lane >> 4;                       // half-wave group

    const int rowBase = blockIdx.x * 128 + wave * 32;
    const int n0 = rowBase + m;                       // A tile 0 row
    const int n1 = rowBase + 16 + m;                  // A tile 1 row
    const int b  = n0 >> 10;                          // 128 | 1024 -> same image
    const float* src0 = inp + (size_t)b * DHW + (n0 & 1023);
    const float* src1 = inp + (size_t)b * DHW + (n1 & 1023);

    // --- load both A tiles once (16-bit A 16x32, contiguous k order, same as B):
    //     half j -> K = 16g + j, within 32-wide chunk
    v16h a0[8], a1[8];
    #pragma unroll
    for (int c0 = 0; c0 < 8; ++c0) {
        #pragma unroll
        for (int j = 0; j < 16; ++j) {
            int d = 32 * c0 + 16 * g + j;
            float x0 = src0[(size_t)d * HW];
            float x1 = src1[(size_t)d * HW];
            x0 = fminf(10.f, fmaxf(-10.f, x0));        // reference clamp
            x1 = fminf(10.f, fmaxf(-10.f, x1));
            a0[c0][j] = (_Float16)x0;
            a1[c0][j] = (_Float16)x1;
        }
    }

    float best0[8], best1[8];
    int   bidx0[8], bidx1[8];
    #pragma unroll
    for (int r = 0; r < 8; ++r) {
        best0[r] = 3.0e38f; bidx0[r] = 0;
        best1[r] = 3.0e38f; bidx1[r] = 0;
    }

#if VQ_ASYNC
    stage_async(embF16, Bsh0, tid);                   // prefetch tile 0
    __builtin_amdgcn_s_wait_asynccnt(0);
    __syncthreads();
#endif

    for (int kt = 0; kt < K_CODES; kt += 16) {
        const _Float16* cur;
#if VQ_ASYNC
        const bool odd = (kt >> 4) & 1;
        cur = odd ? Bsh1 : Bsh0;
        _Float16* nxt = odd ? Bsh0 : Bsh1;
        if (kt + 16 < K_CODES)                        // fill next tile async
            stage_async(embF16 + (size_t)(kt + 16) * D_DIM, nxt, tid);
#else
        cur = Bsh0;
        __syncthreads();
        {   // synchronous fallback staging (16 B per thread per step)
            const uint4* gsrc = (const uint4*)(embF16 + (size_t)kt * D_DIM);
            uint4* ldst = (uint4*)Bsh0;
            #pragma unroll
            for (int ii = 0; ii < 4; ++ii)
                ldst[tid + 128 * ii] = gsrc[tid + 128 * ii];
        }
        __syncthreads();
#endif

        v8f c0acc = {0.f, 0.f, 0.f, 0.f, 0.f, 0.f, 0.f, 0.f};
        v8f c1acc = {0.f, 0.f, 0.f, 0.f, 0.f, 0.f, 0.f, 0.f};
        #pragma unroll
        for (int c0 = 0; c0 < 8; ++c0) {
            // B 32x16, contiguous k order as A: lane = column (codeword m), K = 16g + j
            const _Float16* bp = cur + (size_t)m * D_DIM + 32 * c0 + 16 * g;
            v16h bf;
            #pragma unroll
            for (int j = 0; j < 16; ++j) bf[j] = bp[j];
            c0acc = __builtin_amdgcn_wmma_f32_16x16x32_f16(
                        false, a0[c0], false, bf, (short)0, c0acc, false, false);
            c1acc = __builtin_amdgcn_wmma_f32_16x16x32_f16(
                        false, a1[c0], false, bf, (short)0, c1acc, false, false);
        }

        const float en   = eNorm[kt + m];
        const int   kcol = kt + m;
        #pragma unroll
        for (int r = 0; r < 8; ++r) {
            float d0 = fmaf(-2.f, c0acc[r], en);
            float d1 = fmaf(-2.f, c1acc[r], en);
            if (d0 < best0[r]) { best0[r] = d0; bidx0[r] = kcol; }  // keeps first min
            if (d1 < best1[r]) { best1[r] = d1; bidx1[r] = kcol; }
        }

#if VQ_ASYNC
        __builtin_amdgcn_s_wait_asynccnt(0);          // next tile landed
        __syncthreads();                              // visible to all waves
#endif
    }

    // min+index reduce across the 16 lanes of each half-wave
    #pragma unroll
    for (int off = 1; off < 16; off <<= 1) {
        #pragma unroll
        for (int r = 0; r < 8; ++r) {
            float od0 = __shfl_xor(best0[r], off, 32);
            int   oi0 = __shfl_xor(bidx0[r], off, 32);
            if (od0 < best0[r] || (od0 == best0[r] && oi0 < bidx0[r])) {
                best0[r] = od0; bidx0[r] = oi0;
            }
            float od1 = __shfl_xor(best1[r], off, 32);
            int   oi1 = __shfl_xor(bidx1[r], off, 32);
            if (od1 < best1[r] || (od1 == best1[r] && oi1 < bidx1[r])) {
                best1[r] = od1; bidx1[r] = oi1;
            }
        }
    }

    if (m == 0) {                                      // lanes 0 and 16 write
        #pragma unroll
        for (int r = 0; r < 8; ++r) {
            int row0 = rowBase + r + 8 * g;            // C layout: M = r + 8g
            int row1 = rowBase + 16 + r + 8 * g;
            idxOut[row0]  = bidx0[r];
            idxOut[row1]  = bidx1[r];
            idxOutF[row0] = (float)bidx0[r];
            idxOutF[row1] = (float)bidx1[r];
            atomicAdd(&counts[bidx0[r]], 1u);
            atomicAdd(&counts[bidx1[r]], 1u);
        }
    }
}

// ---------------------------------------------------------------------------
// Kernel 3: gather exact f32 codewords -> quantized output (coalesced along
// hw) + per-block squared-error partial sums (deterministic order).
// grid = (N/256, D); part index = d*128 + nblk.
// ---------------------------------------------------------------------------
__global__ __launch_bounds__(256)
void vq_quant(const float* __restrict__ inp, const float* __restrict__ emb,
              const int* __restrict__ idx, float* __restrict__ outQ,
              float* __restrict__ lossPart) {
    const int d  = blockIdx.y;
    const int nn = blockIdx.x * 256 + threadIdx.x;
    const int b  = nn >> 10;
    const int hw = nn & 1023;
    const size_t off = (size_t)b * DHW + (size_t)d * HW + hw;

    float q = emb[(size_t)d * K_CODES + idx[nn]];      // L2-resident gather
    float x = inp[off];
    outQ[off] = q;                                     // == quantized_st value
    float diff = q - x;

    __shared__ float red[256];
    red[threadIdx.x] = diff * diff;
    __syncthreads();
    for (int s = 128; s > 0; s >>= 1) {
        if (threadIdx.x < s) red[threadIdx.x] += red[threadIdx.x + s];
        __syncthreads();
    }
    if (threadIdx.x == 0)
        lossPart[blockIdx.y * gridDim.x + blockIdx.x] = red[0];
}

// ---------------------------------------------------------------------------
// Kernel 4: finalize loss = 1.25 * mean((q-x)^2) and perplexity.
// Single block, fixed reduction order -> deterministic across replays.
// ---------------------------------------------------------------------------
__global__ __launch_bounds__(256)
void vq_final(const float* __restrict__ lossPart, const unsigned* __restrict__ counts,
              float* __restrict__ out) {
    __shared__ float red[256];
    const int t = threadIdx.x;

    float s = 0.f;
    for (int i = t; i < 32768; i += 256) s += lossPart[i];
    red[t] = s; __syncthreads();
    for (int st = 128; st > 0; st >>= 1) {
        if (t < st) red[t] += red[t + st];
        __syncthreads();
    }
    float lossSum = red[0];
    __syncthreads();

    float h = 0.f;
    for (int k = t; k < K_CODES; k += 256) {
        float p = (float)counts[k] * (1.0f / (float)N_VEC);
        h += p * logf(p + 1e-10f);
    }
    red[t] = h; __syncthreads();
    for (int st = 128; st > 0; st >>= 1) {
        if (t < st) red[t] += red[t + st];
        __syncthreads();
    }
    if (t == 0) {
        out[Q_ELEMS]     = 1.25f * lossSum / (float)Q_ELEMS;  // q_loss + 0.25*e_loss
        out[Q_ELEMS + 1] = expf(-red[0]);                     // perplexity
    }
}

// ---------------------------------------------------------------------------
extern "C" void kernel_launch(void* const* d_in, const int* in_sizes, int n_in,
                              void* d_out, int out_size, void* d_ws, size_t ws_size,
                              hipStream_t stream) {
    const float* inp = (const float*)d_in[0];   // (32, 256, 32, 32) f32
    const float* emb = (const float*)d_in[1];   // (256, 2048) f32
    float* out = (float*)d_out;                 // [quantized | loss | perp | idx]

    char* ws = (char*)d_ws;
    float*    eNorm    = (float*)(ws);                       //   8 KB
    unsigned* counts   = (unsigned*)(ws + 8192);             //   8 KB
    int*      idxI     = (int*)(ws + 16384);                 // 128 KB
    float*    lossPart = (float*)(ws + 16384 + 131072);      // 128 KB
    _Float16* embF16   = (_Float16*)(ws + 16384 + 262144);   //   1 MB (16B aligned)

    vq_prep  <<<8, 256, 0, stream>>>(emb, eNorm, embF16, counts);
    vq_argmin<<<N_VEC / 128, 128, 0, stream>>>(inp, embF16, eNorm, idxI,
                                               out + Q_ELEMS + 2, counts);
    vq_quant <<<dim3(N_VEC / 256, D_DIM), 256, 0, stream>>>(inp, emb, idxI,
                                                            out, lossPart);
    vq_final <<<1, 256, 0, stream>>>(lossPart, counts, out);
}
